// DISCOBlock_27582279975306
// MI455X (gfx1250) — matrix operand order repaired and checked by reading, not compiled, not run on hardware
//
#include <hip/hip_runtime.h>
#include <hip/hip_bf16.h>

// ---------------- static problem config (matches reference) ----------------
#define BATCH   4
#define CH      32          // in and out channels for both convs
#define HH      256
#define WW      256
#define KS      36          // basis count
#define KW      11          // kernel width (2R+1)
#define RAD     5
#define NTAPS   (KW*KW)     // 121
#define TILE_W  128         // output pixels per row-segment per workgroup
#define TILE_H  2           // output rows per workgroup
#define LROWS   (KW + TILE_H - 1)    // 12 staged input rows
#define COLS    (TILE_W + 2*RAD)     // 138 halo columns
#define COLPAD  144
// swizzled kernel: [conv][tap][mtile(2)][lane(32)][16 bf16]
#define KERN_TAP_STRIDE   1024            // ushorts per tap (2*32*16)
#define KERN_CONV_STRIDE  (NTAPS*KERN_TAP_STRIDE)  // 123904 ushorts

typedef __attribute__((ext_vector_type(16))) __bf16    v16bf;
typedef __attribute__((ext_vector_type(8)))  float     v8f;
typedef __attribute__((ext_vector_type(8)))  unsigned  v8u;

static __device__ __forceinline__ unsigned short f32_bf16(float f) {
  unsigned u = __builtin_bit_cast(unsigned, f);
  u += 0x7FFFu + ((u >> 16) & 1u);       // round-to-nearest-even
  return (unsigned short)(u >> 16);
}

// ---------------- fixed DISCO basis PSI[36][121] ----------------
__global__ void k_psi(float* __restrict__ psi) {
  int idx = blockIdx.x * blockDim.x + threadIdx.x;
  if (idx >= KS * NTAPS) return;
  int k = idx / NTAPS, tap = idx % NTAPS;
  int iy = tap / KW - RAD, ix = tap % KW - RAD;
  const float h = 1.0f / 255.0f, cutoff = 0.02f;
  const float PI  = 3.14159265358979323846f;
  const float PI2 = 6.28318530717958647692f;
  float y = iy * h, x = ix * h;
  float r = sqrtf(x * x + y * y);
  float phi = atan2f(y, x); if (phi < 0.f) phi += PI2;
  float inside = (r <= cutoff) ? 1.f : 0.f;
  const float dr = cutoff / 5.f;          // NR-1 = 5
  const float dphi = PI2 / 7.f;           // NPHI = 7
  float val;
  if (k == 0) {
    val = fmaxf(0.f, 1.f - r / dr) * inside;
  } else {
    int j  = 1 + (k - 1) / 7;
    int kk = (k - 1) % 7;
    float rad = fmaxf(0.f, 1.f - fabsf(r - (float)j * dr) / dr) * inside;
    float a = phi - (float)kk * dphi + PI;
    float m = fmodf(a, PI2); if (m < 0.f) m += PI2;   // numpy-style mod
    float d = fabsf(m - PI);
    val = rad * fmaxf(0.f, 1.f - d / dphi);
  }
  psi[idx] = val;
}

// ---------------- build WMMA-swizzled bf16 kernels ----------------
// A-operand lane layout built here for V_WMMA_*_16x16x32 (16-bit A, wave32), contiguous K:
//   lane<16 : M=lane,    VGPR v holds K = {2v,2v+1} (v<8)
//   lane>=16: M=lane-16, same pattern with K += 16
__global__ void k_kern(const float* __restrict__ w1, const float* __restrict__ w2,
                       const float* __restrict__ psi, unsigned short* __restrict__ kern) {
  int idx = blockIdx.x * blockDim.x + threadIdx.x;
  if (idx >= 2 * KERN_CONV_STRIDE) return;
  int v     = idx & 15;
  int lane  = (idx >> 4) & 31;
  int mtile = (idx >> 9) & 1;
  int tap   = (idx >> 10) % NTAPS;
  int conv  = idx / KERN_CONV_STRIDE;
  int vg = v >> 1, pos = v & 1;
  int kch = (vg * 2 + pos) + ((lane & 16) ? 16 : 0);
  int m   = (lane & 15) + mtile * 16;
  const float* w = conv ? w2 : w1;
  float acc = 0.f;
#pragma unroll 6
  for (int kb = 0; kb < KS; ++kb)
    acc += w[(m * CH + kch) * KS + kb] * psi[kb * NTAPS + tap];
  kern[idx] = f32_bf16(acc);
}

// ---------------- misc small kernels ----------------
__global__ void k_zero(float* __restrict__ p, int n) {
  int i = blockIdx.x * blockDim.x + threadIdx.x;
  if (i < n) p[i] = 0.f;
}

// fp32 NCHW image -> bf16 NHWC (coalesced writes)
__global__ void k_cvt(const float* __restrict__ img, unsigned short* __restrict__ dst) {
  int idx = blockIdx.x * blockDim.x + threadIdx.x;     // NHWC linear index
  if (idx >= BATCH * CH * HH * WW) return;
  int c = idx & 31, x = (idx >> 5) & 255, y = (idx >> 13) & 255, b = idx >> 21;
  dst[idx] = f32_bf16(img[((b * CH + c) * HH + y) * WW + x]);
}

// ---------------- implicit-GEMM conv via WMMA ----------------
// One workgroup: batch b, output rows {y, y+1}, 128-pixel segment; 8 waves.
// Each wave computes a 2-row x 16-pixel x 32-channel patch: per tap the two
// A-frags (preswizzled global, L2-resident) feed FOUR WMMAs (B-frags for the
// two rows are adjacent LDS rows) -> 1 global b128 per WMMA, 2x A reuse.
__global__ __launch_bounds__(256)
void k_conv(const unsigned short* __restrict__ in_nhwc,
            const unsigned short* __restrict__ kern,   // pre-offset for this conv
            float* __restrict__ yout,
            float* __restrict__ gsum, float* __restrict__ gsq) {
  __shared__ __align__(64) unsigned short lds_in[LROWS][COLPAD][CH];
  __shared__ float s_sum[CH], s_sq[CH];

  int tid = threadIdx.x;
  int blk = blockIdx.x;                 // b*256 + pair*2 + xt
  int xt   = blk & 1;
  int pair = (blk >> 1) & 127;
  int b    = blk >> 8;
  int yrow = pair * TILE_H;
  int x0   = xt * TILE_W;

  if (tid < CH) { s_sum[tid] = 0.f; s_sq[tid] = 0.f; }

  // stage 12x138 halo, 32 channels (64B) per site, zero-fill borders
  for (int i = tid; i < LROWS * COLS; i += 256) {
    int dy = i / COLS, col = i % COLS;
    int gy = yrow + dy - RAD, gx = x0 + col - RAD;
    uint4* dst = (uint4*)&lds_in[dy][col][0];
    if (gy >= 0 && gy < HH && gx >= 0 && gx < WW) {
      const uint4* src = (const uint4*)&in_nhwc[(((b * HH) + gy) * WW + gx) * CH];
      dst[0] = src[0]; dst[1] = src[1]; dst[2] = src[2]; dst[3] = src[3];
    } else {
      uint4 z = make_uint4(0u, 0u, 0u, 0u);
      dst[0] = z; dst[1] = z; dst[2] = z; dst[3] = z;
    }
  }
  __syncthreads();

  int wave  = tid >> 5, lane = tid & 31;
  int nloc  = lane & 15;          // N (pixel) within tile
  int khalf = lane & 16;          // B-frag channel half: K base 0 or 16
  v8f acc00 = {0.f,0.f,0.f,0.f,0.f,0.f,0.f,0.f};   // row0, ch 0-15
  v8f acc01 = {0.f,0.f,0.f,0.f,0.f,0.f,0.f,0.f};   // row0, ch 16-31
  v8f acc10 = {0.f,0.f,0.f,0.f,0.f,0.f,0.f,0.f};   // row1, ch 0-15
  v8f acc11 = {0.f,0.f,0.f,0.f,0.f,0.f,0.f,0.f};   // row1, ch 16-31

  for (int dy = 0; dy < KW; ++dy) {
#pragma unroll
    for (int dx = 0; dx < KW; ++dx) {
      int tap = dy * KW + dx;
      const unsigned short* kt = kern + tap * KERN_TAP_STRIDE + lane * 16;
      v8u a0raw = *(const v8u*)kt;                         // mtile 0 (out-ch 0-15)
      v8u a1raw = *(const v8u*)(kt + 512);                 // mtile 1 (out-ch 16-31)
      v8u b0raw = *(const v8u*)&lds_in[dy    ][wave * 16 + nloc + dx][khalf];
      v8u b1raw = *(const v8u*)&lds_in[dy + 1][wave * 16 + nloc + dx][khalf];
      __builtin_prefetch(kt + KERN_TAP_STRIDE, 0, 0);      // next tap -> global_prefetch
      v16bf a0 = __builtin_bit_cast(v16bf, a0raw);
      v16bf a1 = __builtin_bit_cast(v16bf, a1raw);
      v16bf b0 = __builtin_bit_cast(v16bf, b0raw);
      v16bf b1 = __builtin_bit_cast(v16bf, b1raw);
      acc00 = __builtin_amdgcn_wmma_f32_16x16x32_bf16(false, a0, false, b0,
                                                      (short)0, acc00, false, false);
      acc01 = __builtin_amdgcn_wmma_f32_16x16x32_bf16(false, a1, false, b0,
                                                      (short)0, acc01, false, false);
      acc10 = __builtin_amdgcn_wmma_f32_16x16x32_bf16(false, a0, false, b1,
                                                      (short)0, acc10, false, false);
      acc11 = __builtin_amdgcn_wmma_f32_16x16x32_bf16(false, a1, false, b1,
                                                      (short)0, acc11, false, false);
    }
  }

  // epilogue: store fp32 NCHW (2 rows) + per-(b,channel) sum/sumsq reduction
  int xp    = x0 + wave * 16 + nloc;
  int mhalf = (lane >> 4) * 8;    // C/D layout: VGPR j -> M = j + 8*(lane/16)
#pragma unroll
  for (int j = 0; j < 8; ++j) {
    int m = mhalf + j;
    float v00 = acc00[j], v01 = acc01[j];
    float v10 = acc10[j], v11 = acc11[j];
    yout[(((b * CH) + m)      * HH + yrow    ) * WW + xp] = v00;
    yout[(((b * CH) + 16 + m) * HH + yrow    ) * WW + xp] = v01;
    yout[(((b * CH) + m)      * HH + yrow + 1) * WW + xp] = v10;
    yout[(((b * CH) + 16 + m) * HH + yrow + 1) * WW + xp] = v11;
    float s0 = v00 + v10, q0 = v00 * v00 + v10 * v10;   // channel m
    float s1 = v01 + v11, q1 = v01 * v01 + v11 * v11;   // channel 16+m
#pragma unroll
    for (int off = 1; off < 16; off <<= 1) {   // reduce over the 16 N-lanes
      s0 += __shfl_xor(s0, off, 32);
      q0 += __shfl_xor(q0, off, 32);
      s1 += __shfl_xor(s1, off, 32);
      q1 += __shfl_xor(q1, off, 32);
    }
    if (nloc == 0) {
      atomicAdd(&s_sum[m], s0);       atomicAdd(&s_sq[m], q0);
      atomicAdd(&s_sum[16 + m], s1);  atomicAdd(&s_sq[16 + m], q1);
    }
  }
  __syncthreads();
  if (tid < CH) {
    atomicAdd(&gsum[b * CH + tid], s_sum[tid]);
    atomicAdd(&gsq [b * CH + tid], s_sq [tid]);
  }
}

// ---------------- InstanceNorm + LeakyReLU ----------------
// mode 0: write bf16 NHWC (feeds next conv); mode 1: write fp32 NCHW (d_out)
__global__ void k_norm(const float* __restrict__ y,
                       const float* __restrict__ gsum, const float* __restrict__ gsq,
                       unsigned short* __restrict__ out_bf, float* __restrict__ out_f,
                       int mode) {
  int idx = blockIdx.x * blockDim.x + threadIdx.x;     // NCHW linear
  if (idx >= BATCH * CH * HH * WW) return;
  int x = idx & 255, yp = (idx >> 8) & 255, c = (idx >> 16) & 31, b = idx >> 21;
  const float inv = 1.0f / (float)(HH * WW);
  float mean = gsum[b * CH + c] * inv;
  float var  = gsq [b * CH + c] * inv - mean * mean;
  float xn = (y[idx] - mean) * rsqrtf(var + 1e-5f);
  float r  = xn >= 0.f ? xn : 0.2f * xn;
  if (mode == 0) out_bf[(((b * HH) + yp) * WW + x) * CH + c] = f32_bf16(r);
  else           out_f[idx] = r;
}

// ---------------- host-side orchestration ----------------
static inline size_t align256(size_t v) { return (v + 255) & ~(size_t)255; }

extern "C" void kernel_launch(void* const* d_in, const int* in_sizes, int n_in,
                              void* d_out, int out_size, void* d_ws, size_t ws_size,
                              hipStream_t stream) {
  const float* image = (const float*)d_in[0];   // [4,32,256,256] f32
  const float* w1    = (const float*)d_in[1];   // [32,32,36] f32
  const float* w2    = (const float*)d_in[2];   // [32,32,36] f32
  float* out = (float*)d_out;                   // [4,32,256,256] f32

  char* ws = (char*)d_ws;
  size_t off = 0;
  float*          psi    = (float*)(ws + off);          off += align256(KS * NTAPS * sizeof(float));
  unsigned short* kern   = (unsigned short*)(ws + off); off += align256((size_t)2 * KERN_CONV_STRIDE * sizeof(unsigned short));
  float*          stats  = (float*)(ws + off);          off += align256(512 * sizeof(float)); // gsum1,gsq1,gsum2,gsq2
  unsigned short* img_bf = (unsigned short*)(ws + off); off += align256((size_t)BATCH * HH * WW * CH * sizeof(unsigned short));
  float*          ybuf   = (float*)(ws + off);          off += align256((size_t)BATCH * CH * HH * WW * sizeof(float));
  unsigned short* h_bf   = img_bf;   // overlay: image consumed before h is produced

  const int NEL = BATCH * CH * HH * WW;          // 8,388,608
  const int TPB = 256;

  k_psi <<<(KS * NTAPS + TPB - 1) / TPB, TPB, 0, stream>>>(psi);
  k_zero<<<2, TPB, 0, stream>>>(stats, 512);
  k_cvt <<<NEL / TPB, TPB, 0, stream>>>(image, img_bf);
  k_kern<<<(2 * KERN_CONV_STRIDE) / TPB, TPB, 0, stream>>>(w1, w2, psi, kern);

  dim3 cgrid(BATCH * (HH / TILE_H) * (WW / TILE_W));   // 1024 workgroups

  // conv1 + stats -> norm1 (bf16 NHWC h)
  k_conv<<<cgrid, TPB, 0, stream>>>(img_bf, kern, ybuf, stats + 0, stats + 128);
  k_norm<<<NEL / TPB, TPB, 0, stream>>>(ybuf, stats + 0, stats + 128, h_bf, nullptr, 0);

  // conv2 + stats -> norm2 (fp32 NCHW final)
  k_conv<<<cgrid, TPB, 0, stream>>>(h_bf, kern + KERN_CONV_STRIDE, ybuf, stats + 256, stats + 384);
  k_norm<<<NEL / TPB, TPB, 0, stream>>>(ybuf, stats + 256, stats + 384, nullptr, out, 1);
}
